// GAT_34832184770812
// MI455X (gfx1250) — matrix operand order repaired and checked by reading, not compiled, not run on hardware
//
#include <hip/hip_runtime.h>
#include <math.h>

#define NN 10000          // nodes
#define NE 320000         // edges
#define INF_ 256          // in feats
#define HID 256           // heads*units
#define NH 4
#define ND 64
#define OUTF 47
#define OUTP 48           // padded
#define NEG 0.2f

typedef __attribute__((ext_vector_type(16))) _Float16 v16h;
typedef __attribute__((ext_vector_type(8)))  _Float16 v8h;
typedef __attribute__((ext_vector_type(8)))  float    v8f;

__device__ __forceinline__ void atomicMaxF(float* a, float v) {
    if (v >= 0.f) atomicMax((int*)a, __float_as_int(v));
    else          atomicMin((unsigned int*)a, __float_as_uint(v));
}

// ---------------- utility kernels ----------------
__global__ void gat_fill(float* p, float v, int n) {
    int i = blockIdx.x * blockDim.x + threadIdx.x;
    if (i < n) p[i] = v;
}

__global__ void gat_cvt_h16(const float* __restrict__ in, _Float16* __restrict__ out, int n) {
    int i = blockIdx.x * blockDim.x + threadIdx.x;
    if (i < n) out[i] = (_Float16)in[i];
}

// pack W [K x Ncols] row-major into per-lane WMMA B-fragment order:
// idx = ((nt*8 + kt)*32 + lane)*16 + j  ->  W[(kt*32 + (lane>=16?16:0) + j)][nt*16 + (lane&15)]
__global__ void gat_packB(const float* __restrict__ W, _Float16* __restrict__ Wp,
                          int ntiles, int ncols) {
    int i = blockIdx.x * blockDim.x + threadIdx.x;
    int total = ntiles * 8 * 32 * 16;
    if (i >= total) return;
    int j    = i & 15;
    int lane = (i >> 4) & 31;
    int kt   = (i >> 9) & 7;
    int nt   = i >> 12;
    int k = kt * 32 + ((lane >= 16) ? 16 : 0) + j;
    int n = nt * 16 + (lane & 15);
    float v = (n < ncols) ? W[(size_t)k * ncols + n] : 0.f;
    Wp[i] = (_Float16)v;
}

// ---------------- WMMA GEMM: C[M x ldc] = A[M x 256] * B ----------------
// one wave per 16x16 tile, K = 256 -> 8 x (16x16x32) WMMA
__global__ void gat_gemm_wmma(const _Float16* __restrict__ A,
                              const _Float16* __restrict__ Bp,
                              float* __restrict__ C,
                              int ntilesN, int ldc) {
    int tile = blockIdx.x;
    int mt = tile / ntilesN;
    int nt = tile - mt * ntilesN;
    int lane = threadIdx.x & 31;
    int arow  = mt * 16 + (lane & 15);
    int kHalf = lane >> 4;            // 0: K {0..15}   1: K {16..31}
    const _Float16* bbase = Bp + ((size_t)nt * 8) * 512 + (size_t)lane * 16;
    v8f c = {};
#pragma unroll
    for (int kt = 0; kt < 8; ++kt) {
        const _Float16* ab = A + (size_t)arow * 256 + kt * 32 + kHalf * 16;
        union { v16h v; v8h h[2]; } a;
        a.h[0] = *(const v8h*)(ab);
        a.h[1] = *(const v8h*)(ab + 8);
        v16h b = *(const v16h*)(bbase + (size_t)kt * 512);
        c = __builtin_amdgcn_wmma_f32_16x16x32_f16(false, a.v, false, b,
                                                   (short)0, c, false, false);
    }
    // C/D layout: lane<16 -> rows 0..7, n=lane ; lane>=16 -> rows 8..15, n=lane-16
    int n = nt * 16 + (lane & 15);
    int rbase = mt * 16 + (lane >> 4) * 8;
#pragma unroll
    for (int r = 0; r < 8; ++r)
        C[(size_t)(rbase + r) * ldc + n] = c[r];
}

// ---------------- layer-1 attention ----------------
__global__ void gat_elr1(const float* __restrict__ feat,
                         const float* __restrict__ al, const float* __restrict__ ar,
                         float* __restrict__ el, float* __restrict__ er) {
    int i = blockIdx.x * blockDim.x + threadIdx.x;   // n*4+h
    if (i >= NN * NH) return;
    int n = i >> 2, h = i & 3;
    const float* f = feat + (size_t)n * HID + h * ND;
    const float* A = al + h * ND;
    const float* B = ar + h * ND;
    float sl = 0.f, sr = 0.f;
#pragma unroll 4
    for (int d = 0; d < ND; ++d) { float v = f[d]; sl += v * A[d]; sr += v * B[d]; }
    el[i] = sl; er[i] = sr;
}

__global__ void gat_max1(const int* __restrict__ src, const int* __restrict__ dst,
                         const float* __restrict__ el, const float* __restrict__ er,
                         float* __restrict__ m) {
    int i = blockIdx.x * blockDim.x + threadIdx.x;
    if (i >= NE * NH) return;
    int e = i >> 2, h = i & 3;
    float x = el[src[e] * NH + h] + er[dst[e] * NH + h];
    x = x >= 0.f ? x : NEG * x;
    atomicMaxF(&m[dst[e] * NH + h], x);
}

__global__ void gat_fixm(float* m, int n) {
    int i = blockIdx.x * blockDim.x + threadIdx.x;
    if (i < n && m[i] == -INFINITY) m[i] = 0.f;
}

__global__ void gat_den1(const int* __restrict__ src, const int* __restrict__ dst,
                         const float* __restrict__ el, const float* __restrict__ er,
                         const float* __restrict__ m, float* __restrict__ den) {
    int i = blockIdx.x * blockDim.x + threadIdx.x;
    if (i >= NE * NH) return;
    int e = i >> 2, h = i & 3;
    int d = dst[e];
    float x = el[src[e] * NH + h] + er[d * NH + h];
    x = x >= 0.f ? x : NEG * x;
    atomicAdd(&den[d * NH + h], expf(x - m[d * NH + h]));
}

__global__ void gat_agg1(const int* __restrict__ src, const int* __restrict__ dst,
                         const float* __restrict__ el, const float* __restrict__ er,
                         const float* __restrict__ m, const float* __restrict__ den,
                         const float* __restrict__ feat, float* __restrict__ agg) {
    int e = blockIdx.x;
    int t = threadIdx.x;              // 0..255
    int h = t >> 6;
    int s = src[e], d = dst[e];
    float x = el[s * NH + h] + er[d * NH + h];
    x = x >= 0.f ? x : NEG * x;
    float a = expf(x - m[d * NH + h]) / fmaxf(den[d * NH + h], 1e-9f);
    atomicAdd(&agg[(size_t)d * HID + t], a * feat[(size_t)s * HID + t]);
}

__global__ void gat_elu_cvt(const float* __restrict__ agg, _Float16* __restrict__ hH, int n) {
    int i = blockIdx.x * blockDim.x + threadIdx.x;
    if (i >= n) return;
    float x = agg[i];
    hH[i] = (_Float16)(x > 0.f ? x : expm1f(x));
}

// ---------------- layer-2 attention (1 head, 47 feats, padded 48) ----------------
__global__ void gat_elr2(const float* __restrict__ feat,
                         const float* __restrict__ al, const float* __restrict__ ar,
                         float* __restrict__ el, float* __restrict__ er) {
    int n = blockIdx.x * blockDim.x + threadIdx.x;
    if (n >= NN) return;
    const float* f = feat + (size_t)n * OUTP;
    float sl = 0.f, sr = 0.f;
    for (int d = 0; d < OUTF; ++d) { float v = f[d]; sl += v * al[d]; sr += v * ar[d]; }
    el[n] = sl; er[n] = sr;
}

__global__ void gat_max2(const int* __restrict__ src, const int* __restrict__ dst,
                         const float* __restrict__ el, const float* __restrict__ er,
                         float* __restrict__ m) {
    int e = blockIdx.x * blockDim.x + threadIdx.x;
    if (e >= NE) return;
    float x = el[src[e]] + er[dst[e]];
    x = x >= 0.f ? x : NEG * x;
    atomicMaxF(&m[dst[e]], x);
}

__global__ void gat_den2(const int* __restrict__ src, const int* __restrict__ dst,
                         const float* __restrict__ el, const float* __restrict__ er,
                         const float* __restrict__ m, float* __restrict__ den) {
    int e = blockIdx.x * blockDim.x + threadIdx.x;
    if (e >= NE) return;
    int d = dst[e];
    float x = el[src[e]] + er[d];
    x = x >= 0.f ? x : NEG * x;
    atomicAdd(&den[d], expf(x - m[d]));
}

__global__ void gat_agg2(const int* __restrict__ src, const int* __restrict__ dst,
                         const float* __restrict__ el, const float* __restrict__ er,
                         const float* __restrict__ m, const float* __restrict__ den,
                         const float* __restrict__ feat, float* __restrict__ agg) {
    int i = blockIdx.x * blockDim.x + threadIdx.x;
    if (i >= NE * OUTF) return;
    int e = i / OUTF;
    int c = i - e * OUTF;
    int s = src[e], d = dst[e];
    float x = el[s] + er[d];
    x = x >= 0.f ? x : NEG * x;
    float a = expf(x - m[d]) / fmaxf(den[d], 1e-9f);
    atomicAdd(&agg[(size_t)d * OUTP + c], a * feat[(size_t)s * OUTP + c]);
}

__global__ void gat_lsm(const float* __restrict__ agg, float* __restrict__ out) {
    int n = blockIdx.x * blockDim.x + threadIdx.x;
    if (n >= NN) return;
    const float* x = agg + (size_t)n * OUTP;
    float mx = -INFINITY;
    for (int d = 0; d < OUTF; ++d) mx = fmaxf(mx, x[d]);
    float s = 0.f;
    for (int d = 0; d < OUTF; ++d) s += expf(x[d] - mx);
    float ls = logf(s);
    for (int d = 0; d < OUTF; ++d) out[(size_t)n * OUTF + d] = x[d] - mx - ls;
}

// ---------------- launcher ----------------
extern "C" void kernel_launch(void* const* d_in, const int* in_sizes, int n_in,
                              void* d_out, int out_size, void* d_ws, size_t ws_size,
                              hipStream_t stream) {
    (void)in_sizes; (void)n_in; (void)out_size; (void)ws_size;
    const float* features = (const float*)d_in[0];
    const int*   src      = (const int*)d_in[1];
    const int*   dst      = (const int*)d_in[2];
    const float* W1       = (const float*)d_in[3];
    const float* al1      = (const float*)d_in[4];
    const float* ar1      = (const float*)d_in[5];
    const float* W2       = (const float*)d_in[6];
    const float* al2      = (const float*)d_in[7];
    const float* ar2      = (const float*)d_in[8];
    float* out = (float*)d_out;

    char* p = (char*)d_ws;
    auto alloc = [&](size_t bytes) -> void* {
        void* r = (void*)p;
        p += (bytes + 255) & ~(size_t)255;
        return r;
    };
    _Float16* featH = (_Float16*)alloc((size_t)NN * INF_ * 2);
    _Float16* W1p   = (_Float16*)alloc((size_t)16 * 8 * 32 * 16 * 2);
    _Float16* W2p   = (_Float16*)alloc((size_t)3 * 8 * 32 * 16 * 2);
    float* feat1 = (float*)alloc((size_t)NN * HID * 4);
    float* el1   = (float*)alloc((size_t)NN * NH * 4);
    float* er1   = (float*)alloc((size_t)NN * NH * 4);
    float* m1    = (float*)alloc((size_t)NN * NH * 4);
    float* dn1   = (float*)alloc((size_t)NN * NH * 4);
    float* agg1  = (float*)alloc((size_t)NN * HID * 4);
    _Float16* hH = (_Float16*)alloc((size_t)NN * HID * 2);
    float* feat2 = (float*)alloc((size_t)NN * OUTP * 4);
    float* el2   = (float*)alloc((size_t)NN * 4);
    float* er2   = (float*)alloc((size_t)NN * 4);
    float* m2    = (float*)alloc((size_t)NN * 4);
    float* dn2   = (float*)alloc((size_t)NN * 4);
    float* agg2  = (float*)alloc((size_t)NN * OUTP * 4);

    auto G = [](long n, int b) { return (unsigned)((n + b - 1) / b); };
    const float NINF = -INFINITY;

    // init (re-done every call: replay-deterministic)
    gat_fill<<<G(NN*HID,256),256,0,stream>>>(agg1, 0.f, NN*HID);
    gat_fill<<<G(NN*NH,256),256,0,stream>>>(dn1, 0.f, NN*NH);
    gat_fill<<<G(NN*NH,256),256,0,stream>>>(m1, NINF, NN*NH);
    gat_fill<<<G(NN*OUTP,256),256,0,stream>>>(agg2, 0.f, NN*OUTP);
    gat_fill<<<G(NN,256),256,0,stream>>>(dn2, 0.f, NN);
    gat_fill<<<G(NN,256),256,0,stream>>>(m2, NINF, NN);

    // precision conversion + B-fragment packing
    gat_cvt_h16<<<G((long)NN*INF_,256),256,0,stream>>>(features, featH, NN*INF_);
    gat_packB<<<G(16*8*32*16,256),256,0,stream>>>(W1, W1p, 16, HID);
    gat_packB<<<G(3*8*32*16,256),256,0,stream>>>(W2, W2p, 3, OUTF);

    // layer 1: GEMM (WMMA) + attention
    gat_gemm_wmma<<<625*16, 32, 0, stream>>>(featH, W1p, feat1, 16, HID);
    gat_elr1<<<G(NN*NH,256),256,0,stream>>>(feat1, al1, ar1, el1, er1);
    gat_max1<<<G((long)NE*NH,256),256,0,stream>>>(src, dst, el1, er1, m1);
    gat_fixm<<<G(NN*NH,256),256,0,stream>>>(m1, NN*NH);
    gat_den1<<<G((long)NE*NH,256),256,0,stream>>>(src, dst, el1, er1, m1, dn1);
    gat_agg1<<<NE, 256, 0, stream>>>(src, dst, el1, er1, m1, dn1, feat1, agg1);
    gat_elu_cvt<<<G((long)NN*HID,256),256,0,stream>>>(agg1, hH, NN*HID);

    // layer 2: GEMM (WMMA) + attention
    gat_gemm_wmma<<<625*3, 32, 0, stream>>>(hH, W2p, feat2, 3, OUTP);
    gat_elr2<<<G(NN,256),256,0,stream>>>(feat2, al2, ar2, el2, er2);
    gat_max2<<<G(NE,256),256,0,stream>>>(src, dst, el2, er2, m2);
    gat_fixm<<<G(NN,256),256,0,stream>>>(m2, NN);
    gat_den2<<<G(NE,256),256,0,stream>>>(src, dst, el2, er2, m2, dn2);
    gat_agg2<<<G((long)NE*OUTF,256),256,0,stream>>>(src, dst, el2, er2, m2, dn2, feat2, agg2);

    // log-softmax -> d_out
    gat_lsm<<<G(NN,256),256,0,stream>>>(agg2, out);
}
